// GCN_6408091205827
// MI455X (gfx1250) — matrix operand order repaired and checked by reading, not compiled, not run on hardware
//
#include <hip/hip_runtime.h>
#include <hip/hip_bf16.h>
#include <cstdint>

// ---------------------------------------------------------------------------
// GCN forward on MI455X (gfx1250): all GEMMs via v_wmma_f32_16x16x32_bf16,
// A-tiles staged with CDNA5 async-to-LDS DMA, double-buffered LDS pipeline.
//   h1 = relu(adj @ (x  @ W1) + b1)
//   h2 = relu(adj @ (h1 @ W2) + b2)
//   h3 =       adj @ (h2 @ W3) + b3
//   seg = segment_max(h3, groups) ; -inf -> 0 ; * mask ; flatten
//   z   = relu(BN(flat @ fc1_w + fc1_b)) ; out = z @ fc2_w + fc2_b
// ---------------------------------------------------------------------------

typedef __bf16 bf16;
typedef __attribute__((ext_vector_type(8)))  __bf16 v8bf;
typedef __attribute__((ext_vector_type(16))) __bf16 v16bf;
typedef __attribute__((ext_vector_type(8)))  float  v8f;

__device__ __forceinline__ unsigned short f2bf_rne(float f) {
  unsigned u = __float_as_uint(f);
  unsigned r = u + 0x7FFFu + ((u >> 16) & 1u);
  return (unsigned short)(r >> 16);
}

// gfx1250 async DMA: global -> LDS, 16 bytes per lane, tracked by ASYNCcnt.
// LDS destination = low 32 bits of the flat shared address (aperture is in
// the high 32 bits per the CDNA5 aperture rules).
__device__ __forceinline__ void async_ld_lds_b128(const void* lds_dst,
                                                  const void* gsrc) {
  unsigned lo = (unsigned)(uintptr_t)lds_dst;
  asm volatile("global_load_async_to_lds_b128 %0, %1, off"
               :
               : "v"(lo), "v"(gsrc)
               : "memory");
}
__device__ __forceinline__ void wait_asynccnt0() {
  asm volatile("s_wait_asynccnt 0x0" ::: "memory");
}

__global__ void cvt_f32_bf16(const float* __restrict__ in,
                             unsigned short* __restrict__ out, int n) {
  for (int i = blockIdx.x * blockDim.x + threadIdx.x; i < n;
       i += gridDim.x * blockDim.x)
    out[i] = f2bf_rne(in[i]);
}

// ---------------------------------------------------------------------------
// Tiled bf16 GEMM, f32 accumulation via WMMA, double-buffered LDS.
//   C[b] = op(A[b] (MxK bf16, rm) @ B[b] (KxN bf16, rm) + bias)
// BM x BN block tile, BK=32. WM x WN waves (WM*WN == 8, 256 threads).
// A staged by async-to-LDS DMA; B staged through registers with transpose
// (LDS holds B as [n][k] so each lane's fragment is one contiguous 32B load).
// All launch dims divide tiles exactly -> EXEC is all-ones at every WMMA.
// ---------------------------------------------------------------------------
template <int BM, int BN, int WM, int WN, bool RELU, bool OUT_BF16>
__global__ __launch_bounds__(WM* WN * 32) void gemm_bf16(
    const bf16* __restrict__ A, const bf16* __restrict__ B,
    void* __restrict__ Cv, const float* __restrict__ bias, int M, int N, int K,
    long long sA, long long sB, long long sC) {
  constexpr int BK = 32;
  constexpr int TM = BM / (WM * 16);
  constexpr int TN = BN / (WN * 16);
  constexpr int NT = WM * WN * 32;
  constexpr int ACH_TOT = BM * BK / 8;  // 16B chunks in an A tile
  constexpr int BCH_TOT = BK * BN / 8;  // 16B chunks in a B tile
  constexpr int ACH = (ACH_TOT + NT - 1) / NT;
  constexpr int BCH = (BCH_TOT + NT - 1) / NT;
  static_assert(BM % (WM * 16) == 0 && BN % (WN * 16) == 0, "tile mismatch");
  static_assert(ACH_TOT % NT == 0, "A staging must be uniform (async EXEC)");

  __shared__ alignas(64) bf16 As[2][BM][BK];  // row-major [m][k]
  __shared__ alignas(64) bf16 Bs[2][BN][BK];  // transposed [n][k]

  A += (long long)blockIdx.z * sA;
  B += (long long)blockIdx.z * sB;

  const int tid  = threadIdx.x;
  const int lane = tid & 31;
  const int wave = tid >> 5;
  const int wm   = wave % WM;
  const int wn   = wave / WM;
  const int hh   = lane >> 4;  // lane half selects K sub-range (contiguous order)
  const int l16  = lane & 15;
  const int mBlock = blockIdx.y * BM;
  const int nBlock = blockIdx.x * BN;

  // Per-thread staging coordinates (constant across K-steps).
  int aRow[ACH], aCol[ACH];
#pragma unroll
  for (int c = 0; c < ACH; ++c) {
    int id = c * NT + tid;
    aRow[c] = id >> 2;          // BK/8 == 4 chunks per row
    aCol[c] = (id & 3) * 8;
  }
  int bKK[BCH], bNC[BCH];
#pragma unroll
  for (int c = 0; c < BCH; ++c) {
    int id = c * NT + tid;
    bKK[c] = id / (BN / 8);
    bNC[c] = (id % (BN / 8)) * 8;
  }

  auto stageA = [&](int step, int buf) {
#pragma unroll
    for (int c = 0; c < ACH; ++c) {
      const bf16* gp =
          A + (long long)(mBlock + aRow[c]) * K + step * BK + aCol[c];
      async_ld_lds_b128(&As[buf][aRow[c]][aCol[c]], gp);
    }
  };
  auto loadB = [&](int step, v8bf* breg) {
#pragma unroll
    for (int c = 0; c < BCH; ++c) {
      if ((BCH_TOT % NT) != 0 && (c * NT + tid) >= BCH_TOT) continue;
      breg[c] = *(const v8bf*)(B + (long long)(step * BK + bKK[c]) * N +
                               nBlock + bNC[c]);
    }
  };
  auto storeB = [&](int buf, const v8bf* breg) {
#pragma unroll
    for (int c = 0; c < BCH; ++c) {
      if ((BCH_TOT % NT) != 0 && (c * NT + tid) >= BCH_TOT) continue;
#pragma unroll
      for (int e = 0; e < 8; ++e) Bs[buf][bNC[c] + e][bKK[c]] = breg[c][e];
    }
  };

  v8f acc[TM][TN] = {};
  auto compute = [&](int buf) {
    v16bf af[TM], bfm[TN];
#pragma unroll
    for (int i = 0; i < TM; ++i) {
      // A 16x32: half 0 -> K 0..15; half 1 -> K 16..31 (contig)
      const bf16* ap = &As[buf][wm * (TM * 16) + i * 16 + l16][hh * 16];
      af[i] = *(const v16bf*)ap;
    }
#pragma unroll
    for (int j = 0; j < TN; ++j)
      // B 32x16: lanes 0-15 hold K 0..15, lanes 16-31 hold K 16..31 (contig)
      bfm[j] = *(const v16bf*)(&Bs[buf][wn * (TN * 16) + j * 16 + l16][hh * 16]);
#pragma unroll
    for (int i = 0; i < TM; ++i)
#pragma unroll
      for (int j = 0; j < TN; ++j)
        acc[i][j] = __builtin_amdgcn_wmma_f32_16x16x32_bf16(
            false, af[i], false, bfm[j], (short)0, acc[i][j], false, false);
  };

  // ---- software pipeline: prologue stages tile 0 ----
  v8bf breg[BCH];
  stageA(0, 0);
  loadB(0, breg);
  storeB(0, breg);
  wait_asynccnt0();
  __syncthreads();

  const int S = K / BK;
  for (int kk = 0; kk < S; ++kk) {
    const int cur = kk & 1;
    if (kk + 1 < S) {
      stageA(kk + 1, cur ^ 1);   // async DMA overlaps the WMMA block below
      loadB(kk + 1, breg);       // global loads issued early, waited post-WMMA
    }
    compute(cur);
    if (kk + 1 < S) storeB(cur ^ 1, breg);
    wait_asynccnt0();            // own DMA done; barrier covers other waves
    __syncthreads();
  }

  // ---- store: C/D layout -> VGPR r: lanes 0-15 M=r, lanes 16-31 M=r+8 ----
  const long long cBatch = (long long)blockIdx.z * sC;
#pragma unroll
  for (int j = 0; j < TN; ++j) {
    int nCol = nBlock + wn * (TN * 16) + j * 16 + l16;
    float bv = (bias != nullptr) ? bias[nCol] : 0.0f;
#pragma unroll
    for (int i = 0; i < TM; ++i) {
      int mBase = mBlock + wm * (TM * 16) + i * 16 + hh * 8;
#pragma unroll
      for (int r = 0; r < 8; ++r) {
        float v = acc[i][j][r] + bv;
        if (RELU) v = fmaxf(v, 0.0f);
        long long idx = cBatch + (long long)(mBase + r) * N + nCol;
        if (OUT_BF16)
          ((unsigned short*)Cv)[idx] = f2bf_rne(v);
        else
          ((float*)Cv)[idx] = v;
      }
    }
  }
}

// ---------------------------------------------------------------------------
// Segment max over groups (1024 nodes -> 64 segments x 16 ch), -inf->0, *mask.
// float max via monotonic-uint key + LDS atomicMax.
// ---------------------------------------------------------------------------
__global__ __launch_bounds__(256) void seg_max_kernel(
    const float* __restrict__ h3, const int* __restrict__ groups,
    const float* __restrict__ mask, float* __restrict__ flat) {
  __shared__ unsigned int segk[64 * 16];
  const int b = blockIdx.x;
  const int tid = threadIdx.x;
  const unsigned NEGINF_KEY = 0x007FFFFFu;  // ~as_uint(-inf)
  for (int i = tid; i < 64 * 16; i += 256) segk[i] = NEGINF_KEY;
  __syncthreads();
  for (int n = tid; n < 1024; n += 256) {
    int g = groups[b * 1024 + n];
    const float* row = h3 + ((long long)b * 1024 + n) * 16;
#pragma unroll
    for (int c = 0; c < 16; ++c) {
      unsigned u = __float_as_uint(row[c]);
      unsigned key = (u & 0x80000000u) ? ~u : (u | 0x80000000u);
      atomicMax(&segk[g * 16 + c], key);
    }
  }
  __syncthreads();
  for (int i = tid; i < 1024; i += 256) {
    int g = i >> 4;
    unsigned key = segk[i];
    float v = 0.0f;
    if (key != NEGINF_KEY) {
      unsigned u = (key & 0x80000000u) ? (key ^ 0x80000000u) : ~key;
      v = __uint_as_float(u);
    }
    v *= mask[b * 64 + g];
    flat[b * 1024 + i] = v;
  }
}

// z = relu(BN(flat @ fc1_w + fc1_b))   [32 x 512], K=1024  (33 MFLOP, tiny)
__global__ __launch_bounds__(512) void fc1_bn_relu_kernel(
    const float* __restrict__ flat, const float* __restrict__ W,
    const float* __restrict__ bias, const float* __restrict__ gamma,
    const float* __restrict__ beta, const float* __restrict__ mean,
    const float* __restrict__ var, float* __restrict__ z) {
  __shared__ float xrow[1024];
  const int b = blockIdx.x, j = threadIdx.x;
  for (int i = j; i < 1024; i += 512) xrow[i] = flat[b * 1024 + i];
  __syncthreads();
  float acc = bias[j];
  for (int k = 0; k < 1024; ++k) acc += xrow[k] * W[k * 512 + j];
  float zn = (acc - mean[j]) * rsqrtf(var[j] + 1e-5f) * gamma[j] + beta[j];
  z[b * 512 + j] = fmaxf(zn, 0.0f);
}

__global__ void fc2_kernel(const float* __restrict__ z,
                           const float* __restrict__ W,
                           const float* __restrict__ bias,
                           float* __restrict__ out) {
  int t = blockIdx.x * blockDim.x + threadIdx.x;
  if (t >= 32 * 10) return;
  int b = t / 10, j = t % 10;
  float acc = bias[j];
  for (int k = 0; k < 512; ++k) acc += z[b * 512 + k] * W[k * 10 + j];
  out[b * 10 + j] = acc;
}

// ---------------------------------------------------------------------------
extern "C" void kernel_launch(void* const* d_in, const int* in_sizes, int n_in,
                              void* d_out, int out_size, void* d_ws,
                              size_t ws_size, hipStream_t stream) {
  (void)in_sizes; (void)n_in; (void)out_size; (void)ws_size;

  const float* x      = (const float*)d_in[0];
  const float* adj    = (const float*)d_in[1];
  const int*   groups = (const int*)d_in[2];
  const float* mask   = (const float*)d_in[3];
  const float* W1 = (const float*)d_in[4];
  const float* b1 = (const float*)d_in[5];
  const float* W2 = (const float*)d_in[6];
  const float* b2 = (const float*)d_in[7];
  const float* W3 = (const float*)d_in[8];
  const float* b3 = (const float*)d_in[9];
  const float* fc1_w = (const float*)d_in[10];
  const float* fc1_b = (const float*)d_in[11];
  const float* bn_g  = (const float*)d_in[12];
  const float* bn_b  = (const float*)d_in[13];
  const float* bn_m  = (const float*)d_in[14];
  const float* bn_v  = (const float*)d_in[15];
  const float* fc2_w = (const float*)d_in[16];
  const float* fc2_b = (const float*)d_in[17];

  const int Bsz = 32, Nn = 1024;
  const long long NB = (long long)Bsz * Nn;  // 32768 rows

  char* ws = (char*)d_ws;
  size_t off = 0;
  auto take = [&](size_t bytes) -> char* {
    char* p = ws + off;
    off = (off + bytes + 255) & ~(size_t)255;
    return p;
  };
  unsigned short* adjb = (unsigned short*)take((size_t)Bsz * Nn * Nn * 2);
  unsigned short* xb   = (unsigned short*)take((size_t)NB * 128 * 2);
  unsigned short* w1b  = (unsigned short*)take((size_t)128 * 256 * 2);
  unsigned short* w2b  = (unsigned short*)take((size_t)256 * 512 * 2);
  unsigned short* w3b  = (unsigned short*)take((size_t)512 * 16 * 2);
  unsigned short* Sb   = (unsigned short*)take((size_t)NB * 512 * 2);
  unsigned short* Hb   = (unsigned short*)take((size_t)NB * 512 * 2);
  float* h3f  = (float*)take((size_t)NB * 16 * 4);
  float* flat = (float*)take((size_t)Bsz * 64 * 16 * 4);
  float* zb   = (float*)take((size_t)Bsz * 512 * 4);

  // ---- one-time bf16 casts (adj stays L2-resident at 67 MB) ----
  cvt_f32_bf16<<<4096, 256, 0, stream>>>(adj, adjb, Bsz * Nn * Nn);
  cvt_f32_bf16<<<1024, 256, 0, stream>>>(x, xb, (int)(NB * 128));
  cvt_f32_bf16<<<32, 256, 0, stream>>>(W1, w1b, 128 * 256);
  cvt_f32_bf16<<<64, 256, 0, stream>>>(W2, w2b, 256 * 512);
  cvt_f32_bf16<<<16, 256, 0, stream>>>(W3, w3b, 512 * 16);

  // ---- layer 1 ----
  gemm_bf16<128, 128, 2, 4, false, true><<<dim3(2, 256, 1), 256, 0, stream>>>(
      (const bf16*)xb, (const bf16*)w1b, Sb, nullptr, 32768, 256, 128, 0, 0, 0);
  gemm_bf16<128, 128, 2, 4, true, true><<<dim3(2, 8, 32), 256, 0, stream>>>(
      (const bf16*)adjb, (const bf16*)Sb, Hb, b1, 1024, 256, 1024,
      (long long)Nn * Nn, (long long)Nn * 256, (long long)Nn * 256);
  // ---- layer 2 ----
  gemm_bf16<128, 128, 2, 4, false, true><<<dim3(4, 256, 1), 256, 0, stream>>>(
      (const bf16*)Hb, (const bf16*)w2b, Sb, nullptr, 32768, 512, 256, 0, 0, 0);
  gemm_bf16<128, 128, 2, 4, true, true><<<dim3(4, 8, 32), 256, 0, stream>>>(
      (const bf16*)adjb, (const bf16*)Sb, Hb, b2, 1024, 512, 1024,
      (long long)Nn * Nn, (long long)Nn * 512, (long long)Nn * 512);
  // ---- layer 3 (N=16) ----
  gemm_bf16<128, 16, 8, 1, false, true><<<dim3(1, 256, 1), 256, 0, stream>>>(
      (const bf16*)Hb, (const bf16*)w3b, Sb, nullptr, 32768, 16, 512, 0, 0, 0);
  gemm_bf16<128, 16, 8, 1, false, false><<<dim3(1, 8, 32), 256, 0, stream>>>(
      (const bf16*)adjb, (const bf16*)Sb, h3f, b3, 1024, 16, 1024,
      (long long)Nn * Nn, (long long)Nn * 16, (long long)Nn * 16);

  // ---- pooling + head ----
  seg_max_kernel<<<32, 256, 0, stream>>>(h3f, groups, mask, flat);
  fc1_bn_relu_kernel<<<32, 512, 0, stream>>>(flat, fc1_w, fc1_b, bn_g, bn_b,
                                             bn_m, bn_v, zb);
  fc2_kernel<<<1, 320, 0, stream>>>(zb, fc2_w, fc2_b, (float*)d_out);
}
